// ReGNN_61598420959319
// MI455X (gfx1250) — matrix operand order repaired and checked by reading, not compiled, not run on hardware
//
#include <hip/hip_runtime.h>
#include <hip/hip_bf16.h>

typedef _Float16 half_t;
typedef __attribute__((ext_vector_type(8)))  _Float16 v8h;
typedef __attribute__((ext_vector_type(16))) _Float16 v16h;
typedef __attribute__((ext_vector_type(8)))  float    v8f;

#define NN   8192      // nodes
#define NE   524288    // edges
#define L0   64        // input labels
#define FA   128       // layer1 out
#define FB   64        // layer2 out
#define FC   32        // layer3 out

static __device__ __forceinline__ void atom_add_f32(float* p, float v) {
    (void)__hip_atomic_fetch_add(p, v, __ATOMIC_RELAXED, __HIP_MEMORY_SCOPE_AGENT);
}

// ---------------- degree / normalization ----------------
__global__ __launch_bounds__(256) void k_deg_init(int* __restrict__ deg) {
    int i = blockIdx.x * 256 + threadIdx.x;
    deg[i] = 1;                               // self-loop
}

__global__ __launch_bounds__(256) void k_deg_count(const int* __restrict__ dst,
                                                   int* __restrict__ deg) {
    int e = blockIdx.x * 256 + threadIdx.x;
    atomicAdd(&deg[dst[e]], 1);               // integer HW atomic
}

__global__ __launch_bounds__(256) void k_deg_rsqrt(const int* __restrict__ deg,
                                                   float* __restrict__ dinv) {
    int i = blockIdx.x * 256 + threadIdx.x;
    dinv[i] = rsqrtf((float)deg[i]);
}

// ---------------- dense H = relu?(X) @ W ----------------
__global__ __launch_bounds__(256) void k_gemm(const float* __restrict__ X,
                                              const float* __restrict__ W,
                                              float* __restrict__ H,
                                              int reluX, int Fi, int foShift) {
    int idx = blockIdx.x * 256 + threadIdx.x;
    int n = idx >> foShift;
    int j = idx & ((1 << foShift) - 1);
    const float* xr = X + (size_t)n * Fi;
    float acc = 0.f;
    for (int k = 0; k < Fi; ++k) {
        float xv = xr[k];
        if (reluX) xv = fmaxf(xv, 0.f);
        acc = fmaf(xv, W[((size_t)k << foShift) + j], acc);
    }
    H[idx] = acc;
}

// ---------------- G = H * dinv^2 (self-loop) + bias ----------------
__global__ __launch_bounds__(256) void k_agg_init(const float* __restrict__ H,
                                                  const float* __restrict__ dinv,
                                                  const float* __restrict__ bias,
                                                  float* __restrict__ G, int foShift) {
    int idx = blockIdx.x * 256 + threadIdx.x;
    int n = idx >> foShift;
    int j = idx & ((1 << foShift) - 1);
    float di = dinv[n];
    G[idx] = H[idx] * di * di + bias[j];
}

// ---------------- edge scatter: G[dst] += H[src] * dinv[s]*dinv[d] ----------------
__global__ __launch_bounds__(256) void k_scatter(const int* __restrict__ src,
                                                 const int* __restrict__ dst,
                                                 const float* __restrict__ H,
                                                 const float* __restrict__ dinv,
                                                 float* __restrict__ G,
                                                 int perShift, int foShift) {
    int idx = blockIdx.x * 256 + threadIdx.x;
    int e = idx >> perShift;
    int c = (idx & ((1 << perShift) - 1)) << 2;
    int s = src[e], d = dst[e];
    float nm = dinv[s] * dinv[d];
    const float4 v = *(const float4*)(H + ((size_t)s << foShift) + c);
    float* g = G + ((size_t)d << foShift) + c;
    atom_add_f32(g + 0, v.x * nm);
    atom_add_f32(g + 1, v.y * nm);
    atom_add_f32(g + 2, v.z * nm);
    atom_add_f32(g + 3, v.w * nm);
}

// ---------------- column sum of [NN x 32] -> cs[32] ----------------
__global__ __launch_bounds__(256) void k_colsum(const float* __restrict__ F,
                                                float* __restrict__ cs) {
    __shared__ float sm[256];
    int t = threadIdx.x;
    int col = t & 31;
    int rg = t >> 5;                          // 0..7
    int r0 = blockIdx.x * 128 + rg;           // 64 blocks x 128 rows
    float acc = 0.f;
#pragma unroll
    for (int i = 0; i < 16; ++i)
        acc += F[(size_t)(r0 + i * 8) * FC + col];
    sm[t] = acc;
    __syncthreads();
    if (t < 32) {
        float s = 0.f;
#pragma unroll
        for (int i = 0; i < 8; ++i) s += sm[col + i * 32];
        atom_add_f32(cs + col, s);
    }
}

__global__ void k_zero64(float* __restrict__ p) { p[threadIdx.x] = 0.f; }

// m[0..31] = tanh(cs1 @ M1), m[32..63] = tanh(cs2 @ M2)
__global__ void k_match(const float* __restrict__ cs,
                        const float* __restrict__ M1,
                        const float* __restrict__ M2,
                        float* __restrict__ m) {
    int t = threadIdx.x;                      // 0..63
    int j = t & 31;
    const float* c = cs + (t >> 5) * 32;
    const float* M = (t < 32) ? M1 : M2;
    float acc = 0.f;
#pragma unroll
    for (int k = 0; k < 32; ++k) acc = fmaf(c[k], M[k * 32 + j], acc);
    m[t] = tanhf(acc);
}

// ah[n][j] = (f16)(f[n][j] - m[j])
__global__ __launch_bounds__(256) void k_amake(const float* __restrict__ f,
                                               const float* __restrict__ m,
                                               half_t* __restrict__ ah) {
    int idx = blockIdx.x * 256 + threadIdx.x;
    ah[idx] = (half_t)(f[idx] - m[idx & 31]);
}

// ---------------- scores = sigmoid(a1 @ a2^T) via WMMA f16->f32 ----------------
// One v_wmma_f32_16x16x32_f16 per 16x16 output tile (K = 32 exactly).
// A operand (contiguous k, same in B): lane l(0-15) row m=l loads K{0..15}; lane l+16 loads K{16..31}.
// B operand (= a2^T): lane l(0-15) col n=l loads a2h[n][0..15]; lane l+16 -> a2h[n][16..31].
// Epilogue: sigmoid via v_exp_f32 + raw v_rcp_f32; NT stores keep a1h/a2h hot in L2.
__global__ __launch_bounds__(256) void k_final(const half_t* __restrict__ A,
                                               const half_t* __restrict__ B,
                                               float* __restrict__ out) {
    const int lane = threadIdx.x & 31;
    const int wave = threadIdx.x >> 5;        // 8 waves/block
    const int hf   = lane >> 4;
    const int l    = lane & 15;
    const int tm   = blockIdx.x;              // 512 tile rows

    const half_t* ar = A + (size_t)(tm * 16 + l) * FC + hf * 16;
    const v16h av = *(const v16h*)ar;

    const int rowbase = tm * 16 + hf * 8;
    for (int tn = wave * 64; tn < wave * 64 + 64; ++tn) {
        const v16h bv = *(const v16h*)(B + (size_t)(tn * 16 + l) * FC + hf * 16);
        v8f c = {};
        c = __builtin_amdgcn_wmma_f32_16x16x32_f16(
                false, av, false, bv, (short)0, c, false, false);
        float* op = out + (size_t)rowbase * NN + tn * 16 + l;
#pragma unroll
        for (int r = 0; r < 8; ++r) {
            float e = __expf(-c[r]);                       // v_exp_f32
            float s = __builtin_amdgcn_rcpf(1.0f + e);     // v_rcp_f32 (no IEEE div expansion)
            __builtin_nontemporal_store(s, op + (size_t)r * NN);
        }
    }
}

// ---------------- host orchestration ----------------
static void run_layer(const int* src, const int* dst, const float* X,
                      const float* W, const float* bias,
                      float* H, float* G, const float* dinv,
                      int Fi, int foShift, int reluX, hipStream_t st) {
    int total  = NN << foShift;
    int stotal = NE << (foShift - 2);
    k_gemm    <<<total  / 256, 256, 0, st>>>(X, W, H, reluX, Fi, foShift);
    k_agg_init<<<total  / 256, 256, 0, st>>>(H, dinv, bias, G, foShift);
    k_scatter <<<stotal / 256, 256, 0, st>>>(src, dst, H, dinv, G, foShift - 2, foShift);
}

static void run_graph(const int* src, const int* dst, const float* x,
                      const float* W1, const float* b1,
                      const float* W2, const float* b2,
                      const float* W3, const float* b3,
                      float* A, float* Bf, int* deg, float* dinv,
                      float* fout, hipStream_t st) {
    k_deg_init <<<NN / 256, 256, 0, st>>>(deg);
    k_deg_count<<<NE / 256, 256, 0, st>>>(dst, deg);
    k_deg_rsqrt<<<NN / 256, 256, 0, st>>>(deg, dinv);
    run_layer(src, dst, x,  W1, b1, A, Bf,   dinv, L0, 7, 0, st); // -> B [N,128]
    run_layer(src, dst, Bf, W2, b2, A, Bf,   dinv, FA, 6, 1, st); // relu on read; -> B [N,64]
    run_layer(src, dst, Bf, W3, b3, A, fout, dinv, FB, 5, 1, st); // -> fout [N,32], no relu
}

extern "C" void kernel_launch(void* const* d_in, const int* in_sizes, int n_in,
                              void* d_out, int out_size, void* d_ws, size_t ws_size,
                              hipStream_t stream) {
    const int*   ei1 = (const int*)  d_in[0];
    const int*   ei2 = (const int*)  d_in[1];
    const float* x1  = (const float*)d_in[2];
    const float* x2  = (const float*)d_in[3];
    const float* W1  = (const float*)d_in[4];
    const float* b1  = (const float*)d_in[5];
    const float* W2  = (const float*)d_in[6];
    const float* b2  = (const float*)d_in[7];
    const float* W3  = (const float*)d_in[8];
    const float* b3  = (const float*)d_in[9];
    const float* M1  = (const float*)d_in[10];
    const float* M2  = (const float*)d_in[11];
    float* out = (float*)d_out;

    char* w = (char*)d_ws;
    size_t off = 0;
    auto take = [&](size_t bytes) { char* p = w + off; off = (off + bytes + 255) & ~(size_t)255; return p; };
    float* A    = (float*) take((size_t)NN * FA * 4);
    float* Bf   = (float*) take((size_t)NN * FA * 4);
    float* f1   = (float*) take((size_t)NN * FC * 4);
    float* f2   = (float*) take((size_t)NN * FC * 4);
    int*   deg  = (int*)   take((size_t)NN * 4);
    float* dinv = (float*) take((size_t)NN * 4);
    float* cs   = (float*) take(64 * 4);          // cs1 || cs2
    float* m    = (float*) take(64 * 4);          // m1  || m2
    half_t* a1h = (half_t*)take((size_t)NN * FC * 2);
    half_t* a2h = (half_t*)take((size_t)NN * FC * 2);
    (void)ws_size; (void)in_sizes; (void)n_in; (void)out_size;

    // two GCN passes (shared deg/dinv scratch; sequential on stream)
    run_graph(ei1, ei1 + NE, x1, W1, b1, W2, b2, W3, b3, A, Bf, deg, dinv, f1, stream);
    run_graph(ei2, ei2 + NE, x2, W1, b1, W2, b2, W3, b3, A, Bf, deg, dinv, f2, stream);

    // matching vectors: m1 = tanh(colsum(f1)@M1), m2 = tanh(colsum(f2)@M2)
    k_zero64<<<1, 64, 0, stream>>>(cs);
    k_colsum<<<64, 256, 0, stream>>>(f1, cs);
    k_colsum<<<64, 256, 0, stream>>>(f2, cs + 32);
    k_match <<<1, 64, 0, stream>>>(cs, M1, M2, m);

    // a1 = f1 - m2 ; a2 = f2 - m1  (fp32 -> f16 for WMMA)
    k_amake<<<(NN * FC) / 256, 256, 0, stream>>>(f1, m + 32, a1h);
    k_amake<<<(NN * FC) / 256, 256, 0, stream>>>(f2, m,      a2h);

    // scores = sigmoid(a1 @ a2^T), 512 tile-rows x (8 waves x 64 tile-cols)
    k_final<<<512, 256, 0, stream>>>(a1h, a2h, out);
}
